// MultiHeadAttention_21079699489368
// MI455X (gfx1250) — matrix operand order repaired and checked by reading, not compiled, not run on hardware
//
#include <hip/hip_runtime.h>
#include <math.h>

// ---------------------------------------------------------------------------
// MultiHeadAttention forward for MI455X (gfx1250, wave32, WMMA).
// B=2, S=2048, D=512, H=8, DK=DV=64.  Outputs: out [B,S,D] fp32, attn [16,S,S] fp32.
// All GEMMs use v_wmma_f32_16x16x32_bf16 (bf16 inputs, fp32 accumulate).
// GEMM K-loops are ping-pong software-pipelined (unroll-by-2, loads write the
// loop-carried fragment registers directly -> no v_mov rotation, partial waits).
// ---------------------------------------------------------------------------

typedef __attribute__((ext_vector_type(16))) __bf16 v16bf;
typedef __attribute__((ext_vector_type(8)))  float  v8f;

#define S_LEN 2048
#define D_MOD 512
#define NHB   16          // H*B
#define HD    64          // DK == DV

union FragBF { v16bf v; uint4 u[2]; };

__device__ __forceinline__ unsigned short f32_to_bf16(float f) {
  unsigned u = __builtin_bit_cast(unsigned, f);
  unsigned r = 0x7FFFu + ((u >> 16) & 1u);      // round-to-nearest-even
  return (unsigned short)((u + r) >> 16);
}

// A-matrix 16x32 bf16 fragment (ISA 7.12.2): lane&15 = row M, lane>>4 selects
// K half-groups {0-7,16-23} vs {8-15,24-31}. Two contiguous 16B loads at +0/+32B.
__device__ __forceinline__ v16bf load_frag_a(const unsigned short* base, int row0,
                                             int rowlen, int k0, int lane) {
  int row = row0 + (lane & 15);
  int k   = k0 + ((lane >> 4) << 3);            // +8 elements for upper lanes
  const unsigned short* p = base + (size_t)row * rowlen + k;
  FragBF f;
  f.u[0] = *(const uint4*)(p);                  // K = k .. k+7
  f.u[1] = *(const uint4*)(p + 16);             // K = k+16 .. k+23
  return f.v;
}

// B-matrix 32x16 bf16 fragment: source stored row-major as [n][k] ("n" rows hold
// the B column data contiguously in k). lane&15 = column N, lane>>4 selects K
// half-groups {0-7,16-23} vs {8-15,24-31}. Two contiguous 16B loads at +0/+32B.
__device__ __forceinline__ v16bf load_frag_b(const unsigned short* base, int n0,
                                             int rowlen, int k0, int lane) {
  int n = n0 + (lane & 15);
  int k = k0 + ((lane >> 4) << 3);              // +8 elements for upper lanes
  const unsigned short* p = base + (size_t)n * rowlen + k;
  FragBF f;
  f.u[0] = *(const uint4*)(p);                  // K = k .. k+7
  f.u[1] = *(const uint4*)(p + 16);             // K = k+16 .. k+23
  return f.v;
}

__device__ __forceinline__ v8f wmma_bf16(v16bf a, v16bf b, v8f c) {
  return __builtin_amdgcn_wmma_f32_16x16x32_bf16(false, a, false, b,
                                                 (short)0, c, false, false);
}

// ---------------------------------------------------------------------------
// Kernel 0: convert q/k/v and weights to bf16; transpose wq/wk/wv to [h][dk][d].
// ---------------------------------------------------------------------------
__global__ __launch_bounds__(256) void mha_pack_kernel(
    const float* __restrict__ q, const float* __restrict__ k, const float* __restrict__ v,
    const float* __restrict__ wq, const float* __restrict__ wk, const float* __restrict__ wv,
    const float* __restrict__ pw,
    unsigned short* __restrict__ qb, unsigned short* __restrict__ kb, unsigned short* __restrict__ vb,
    unsigned short* __restrict__ wqT, unsigned short* __restrict__ wkT, unsigned short* __restrict__ wvT,
    unsigned short* __restrict__ pwb) {
  const int NQ = 2 * S_LEN * D_MOD;             // 2,097,152
  const int NW = 8 * D_MOD * HD;                // 262,144
  const int NP = D_MOD * D_MOD;                 // 262,144
  int i = blockIdx.x * 256 + threadIdx.x;
  if (i < NQ) { qb[i] = f32_to_bf16(q[i]); kb[i] = f32_to_bf16(k[i]); vb[i] = f32_to_bf16(v[i]); }
  if (i < NW) {
    int h = i >> 15;                            // 512*64 per head
    int r = i & 32767;
    int d = r >> 6, dk = r & 63;
    int o = (h << 15) + dk * D_MOD + d;         // [h][dk][d]
    wqT[o] = f32_to_bf16(wq[i]); wkT[o] = f32_to_bf16(wk[i]); wvT[o] = f32_to_bf16(wv[i]);
  }
  if (i < NP) pwb[i] = f32_to_bf16(pw[i]);      // proj_w [dout][f] already B-friendly
}

// ---------------------------------------------------------------------------
// Kernel 1: QKV projections.  out[s,dk] = sum_d in[b,s,d] * wT[h,dk,d].
// One WG (8 waves) per (which, hb, 128-row block); each wave: 16 rows x 64 cols.
// qh/kh stored row-major [hb][s][64]; vh stored transposed vht[hb][dv][s].
// K loop: ping-pong buffers, unrolled by 2, prefetch index wraps (&15) on the
// final step so there is no conditional (no phi copies, one harmless reload).
// ---------------------------------------------------------------------------
__global__ __launch_bounds__(256) void mha_qkv_proj_kernel(
    const unsigned short* __restrict__ qb, const unsigned short* __restrict__ kb,
    const unsigned short* __restrict__ vb,
    const unsigned short* __restrict__ wqT, const unsigned short* __restrict__ wkT,
    const unsigned short* __restrict__ wvT,
    unsigned short* __restrict__ qh, unsigned short* __restrict__ kh,
    unsigned short* __restrict__ vht) {
  int bx = blockIdx.x;                          // 0..767
  int which = bx >> 8;                          // 0=q 1=k 2=v
  int rem = bx & 255;
  int hb = rem >> 4, sblk = rem & 15;
  int h = hb >> 1, b = hb & 1;
  int tid = threadIdx.x, lane = tid & 31, w = tid >> 5;

  const unsigned short* A  = (which == 0) ? qb  : (which == 1) ? kb  : vb;
  const unsigned short* Wt = (which == 0) ? wqT : (which == 1) ? wkT : wvT;
  const unsigned short* Ab = A  + (size_t)b * S_LEN * D_MOD;
  const unsigned short* Wb = Wt + (size_t)h * HD * D_MOD;

  int r0 = sblk * 128 + w * 16;
  v8f acc[4] = {};

  // buffer 0 primed with kc = 0
  v16bf a0f = load_frag_a(Ab, r0, D_MOD, 0, lane);
  v16bf b0f[4];
#pragma unroll
  for (int j = 0; j < 4; ++j) b0f[j] = load_frag_b(Wb, j * 16, D_MOD, 0, lane);

  for (int kc = 0; kc < 16; kc += 2) {          // K = 512 in chunks of 32, x2 per iter
    // buffer 1 <- kc+1 (always in range)
    v16bf a1f = load_frag_a(Ab, r0, D_MOD, (kc + 1) * 32, lane);
    v16bf b1f[4];
#pragma unroll
    for (int j = 0; j < 4; ++j) b1f[j] = load_frag_b(Wb, j * 16, D_MOD, (kc + 1) * 32, lane);
#pragma unroll
    for (int j = 0; j < 4; ++j) acc[j] = wmma_bf16(a0f, b0f[j], acc[j]);
    // buffer 0 <- kc+2 (wraps to 0 on last iter: redundant, in-bounds, no phi)
    int kn = (kc + 2) & 15;
    a0f = load_frag_a(Ab, r0, D_MOD, kn * 32, lane);
#pragma unroll
    for (int j = 0; j < 4; ++j) b0f[j] = load_frag_b(Wb, j * 16, D_MOD, kn * 32, lane);
#pragma unroll
    for (int j = 0; j < 4; ++j) acc[j] = wmma_bf16(a1f, b1f[j], acc[j]);
  }

  int rsel = (lane >> 4) << 3, c = lane & 15;   // C/D layout: VGPR i, lanes 0-15 -> M=i, 16-31 -> M=i+8
  if (which < 2) {
    unsigned short* O = ((which == 0) ? qh : kh) + (size_t)hb * S_LEN * HD;
#pragma unroll
    for (int j = 0; j < 4; ++j)
#pragma unroll
      for (int i = 0; i < 8; ++i)
        O[(size_t)(r0 + i + rsel) * HD + j * 16 + c] = f32_to_bf16(acc[j][i]);
  } else {
    unsigned short* O = vht + (size_t)hb * HD * S_LEN;
#pragma unroll
    for (int j = 0; j < 4; ++j)
#pragma unroll
      for (int i = 0; i < 8; ++i)
        O[(size_t)(j * 16 + c) * S_LEN + (r0 + i + rsel)] = f32_to_bf16(acc[j][i]);
  }
}

// ---------------------------------------------------------------------------
// Kernel 2: attention per (hb, 128-query-row block); wave owns 16 query rows.
// Pass A: stream K-tiles, WMMA scores, exact online max/sum (scores discarded).
// Pass B: recompute scores, write normalized attn ONCE (268MB = the roofline),
//         transpose P via 1KB/wave LDS stage, accumulate O = P x V with WMMA.
// Next K-tile's B fragments prefetched with a wrapping index (no tail branch).
// ---------------------------------------------------------------------------
__global__ __launch_bounds__(256) void mha_attn_kernel(
    const unsigned short* __restrict__ qh, const unsigned short* __restrict__ kh,
    const unsigned short* __restrict__ vht,
    float* __restrict__ attn, unsigned short* __restrict__ ob) {
  __shared__ __align__(16) unsigned short pstage[8][16][32];   // per-wave P transpose stage

  int bx = blockIdx.x;                          // 0..255
  int hb = bx >> 4, sblk = bx & 15;
  int h = hb >> 1, b = hb & 1;
  int tid = threadIdx.x, lane = tid & 31, w = tid >> 5;
  int q0 = sblk * 128 + w * 16;

  const unsigned short* Q = qh  + (size_t)hb * S_LEN * HD;
  const unsigned short* K = kh  + (size_t)hb * S_LEN * HD;
  const unsigned short* V = vht + (size_t)hb * HD * S_LEN;
  const float scale = 0.044194173824159216f;    // 1/sqrt(512) (ref scales by sqrt(d_model))

  v16bf a0 = load_frag_a(Q, q0, HD, 0,  lane);  // q rows, dk 0..31
  v16bf a1 = load_frag_a(Q, q0, HD, 32, lane);  // q rows, dk 32..63

  float m[8], l[8];
#pragma unroll
  for (int i = 0; i < 8; ++i) { m[i] = -1e30f; l[i] = 0.f; }

  // ---- Pass A: row max & sum of exp (exact, full row) ----
  {
    v16bf b0 = load_frag_b(K, 0, HD, 0,  lane);
    v16bf b1 = load_frag_b(K, 0, HD, 32, lane);
    for (int skt = 0; skt < 128; ++skt) {
      v8f acc = {};
      acc = wmma_bf16(a0, b0, acc);
      acc = wmma_bf16(a1, b1, acc);
      int nxt = ((skt + 1) & 127) * 16;          // wraps on last iter (harmless reload)
      b0 = load_frag_b(K, nxt, HD, 0,  lane);    // in flight over softmax VALU below
      b1 = load_frag_b(K, nxt, HD, 32, lane);
#pragma unroll
      for (int i = 0; i < 8; ++i) {
        float vsc = acc[i] * scale;
        float mx = vsc;                          // reduce within 16-lane row group
        mx = fmaxf(mx, __shfl_xor(mx, 1, 32));
        mx = fmaxf(mx, __shfl_xor(mx, 2, 32));
        mx = fmaxf(mx, __shfl_xor(mx, 4, 32));
        mx = fmaxf(mx, __shfl_xor(mx, 8, 32));
        float nm = fmaxf(m[i], mx);
        float e = __expf(vsc - nm);
        e += __shfl_xor(e, 1, 32);
        e += __shfl_xor(e, 2, 32);
        e += __shfl_xor(e, 4, 32);
        e += __shfl_xor(e, 8, 32);
        l[i] = l[i] * __expf(m[i] - nm) + e;
        m[i] = nm;
      }
    }
  }
  float rl[8];
#pragma unroll
  for (int i = 0; i < 8; ++i) rl[i] = 1.0f / l[i];

  // ---- Pass B: recompute, write attn once, accumulate O = P x V ----
  v8f acco[4] = {};
  float* attn_hb = attn + (size_t)hb * S_LEN * S_LEN;
  int rsel = (lane >> 4) << 3, c = lane & 15;
  {
    v16bf b0 = load_frag_b(K, 0, HD, 0,  lane);
    v16bf b1 = load_frag_b(K, 0, HD, 32, lane);
    for (int skt = 0; skt < 128; ++skt) {
      v8f acc = {};
      acc = wmma_bf16(a0, b0, acc);
      acc = wmma_bf16(a1, b1, acc);
      int nxt = ((skt + 1) & 127) * 16;
      b0 = load_frag_b(K, nxt, HD, 0,  lane);
      b1 = load_frag_b(K, nxt, HD, 32, lane);
#pragma unroll
      for (int i = 0; i < 8; ++i) {
        float p = __expf(acc[i] * scale - m[i]) * rl[i];
        int row = i + rsel;
        attn_hb[(size_t)(q0 + row) * S_LEN + skt * 16 + c] = p;        // the 268MB write
        pstage[w][row][(skt & 1) * 16 + c] = f32_to_bf16(p);           // lane-transpose stage
      }
      if (skt & 1) {                             // every 32 s_k columns: one A frag, 4 WMMAs
        v16bf ap = load_frag_a(&pstage[w][0][0], 0, 32, 0, lane);      // same-wave LDS: in-order
        int sk0 = (skt >> 1) * 32;
#pragma unroll
        for (int j = 0; j < 4; ++j) {
          v16bf bv = load_frag_b(V, j * 16, S_LEN, sk0, lane);         // vht[dv][s] rows
          acco[j] = wmma_bf16(ap, bv, acco[j]);
        }
      }
    }
  }
  // store O head-concat layout: ob[b][s][h*64+dv] (bf16, feeds output projection)
  unsigned short* Ob = ob + (size_t)b * S_LEN * D_MOD;
#pragma unroll
  for (int j = 0; j < 4; ++j)
#pragma unroll
    for (int i = 0; i < 8; ++i)
      Ob[(size_t)(q0 + i + rsel) * D_MOD + h * HD + j * 16 + c] = f32_to_bf16(acco[j][i]);
}

// ---------------------------------------------------------------------------
// Kernel 3: output projection  y[s,dout] = sum_f o[s,f] * proj_w[dout,f].
// Ping-pong K loop like kernel 1.
// ---------------------------------------------------------------------------
__global__ __launch_bounds__(256) void mha_proj_kernel(
    const unsigned short* __restrict__ ob, const unsigned short* __restrict__ pwb,
    float* __restrict__ y) {
  int tid = threadIdx.x, lane = tid & 31, w = tid >> 5;
  int tile = blockIdx.x * 8 + w;                // 2048 wave tiles
  int rb = tile >> 3, db = tile & 7;
  int r0 = rb * 16, n00 = db * 64;
  v8f acc[4] = {};

  v16bf a0f = load_frag_a(ob, r0, D_MOD, 0, lane);
  v16bf b0f[4];
#pragma unroll
  for (int j = 0; j < 4; ++j) b0f[j] = load_frag_b(pwb, n00 + j * 16, D_MOD, 0, lane);

  for (int kc = 0; kc < 16; kc += 2) {
    v16bf a1f = load_frag_a(ob, r0, D_MOD, (kc + 1) * 32, lane);
    v16bf b1f[4];
#pragma unroll
    for (int j = 0; j < 4; ++j) b1f[j] = load_frag_b(pwb, n00 + j * 16, D_MOD, (kc + 1) * 32, lane);
#pragma unroll
    for (int j = 0; j < 4; ++j) acc[j] = wmma_bf16(a0f, b0f[j], acc[j]);
    int kn = (kc + 2) & 15;
    a0f = load_frag_a(ob, r0, D_MOD, kn * 32, lane);
#pragma unroll
    for (int j = 0; j < 4; ++j) b0f[j] = load_frag_b(pwb, n00 + j * 16, D_MOD, kn * 32, lane);
#pragma unroll
    for (int j = 0; j < 4; ++j) acc[j] = wmma_bf16(a1f, b1f[j], acc[j]);
  }

  int rsel = (lane >> 4) << 3, c = lane & 15;
#pragma unroll
  for (int j = 0; j < 4; ++j)
#pragma unroll
    for (int i = 0; i < 8; ++i)
      y[(size_t)(r0 + i + rsel) * D_MOD + n00 + j * 16 + c] = acc[j][i];
}

// ---------------------------------------------------------------------------
// Kernel 4: out = LayerNorm(y + proj_b + q)  (fp32 exact, biased variance).
// One block per row of 512; 256 threads x 2 elements.
// ---------------------------------------------------------------------------
__global__ __launch_bounds__(256) void mha_ln_kernel(
    const float* __restrict__ y, const float* __restrict__ qin,
    const float* __restrict__ pb, const float* __restrict__ g,
    const float* __restrict__ be, float* __restrict__ out) {
  int row = blockIdx.x, t = threadIdx.x;
  int lane = t & 31, w = t >> 5;
  const float* yr = y   + (size_t)row * D_MOD;
  const float* qr = qin + (size_t)row * D_MOD;
  float x0 = yr[t]       + pb[t]       + qr[t];
  float x1 = yr[t + 256] + pb[t + 256] + qr[t + 256];

  __shared__ float red[8];
  float s = x0 + x1;
  for (int d = 1; d < 32; d <<= 1) s += __shfl_xor(s, d, 32);
  if (lane == 0) red[w] = s;
  __syncthreads();
  float tot = red[0] + red[1] + red[2] + red[3] + red[4] + red[5] + red[6] + red[7];
  float mu = tot * (1.0f / 512.0f);
  float d0 = x0 - mu, d1 = x1 - mu;
  __syncthreads();
  float sv = d0 * d0 + d1 * d1;
  for (int d = 1; d < 32; d <<= 1) sv += __shfl_xor(sv, d, 32);
  if (lane == 0) red[w] = sv;
  __syncthreads();
  float vt = red[0] + red[1] + red[2] + red[3] + red[4] + red[5] + red[6] + red[7];
  float rs = rsqrtf(vt * (1.0f / 512.0f) + 1e-5f);
  float* orow = out + (size_t)row * D_MOD;
  orow[t]       = d0 * rs * g[t]       + be[t];
  orow[t + 256] = d1 * rs * g[t + 256] + be[t + 256];
}

// ---------------------------------------------------------------------------
// Launch
// ---------------------------------------------------------------------------
extern "C" void kernel_launch(void* const* d_in, const int* in_sizes, int n_in,
                              void* d_out, int out_size, void* d_ws, size_t ws_size,
                              hipStream_t stream) {
  const float* q  = (const float*)d_in[0];
  const float* k  = (const float*)d_in[1];
  const float* v  = (const float*)d_in[2];
  const float* wq = (const float*)d_in[3];
  const float* wk = (const float*)d_in[4];
  const float* wv = (const float*)d_in[5];
  const float* pw = (const float*)d_in[6];
  const float* pb = (const float*)d_in[7];
  const float* ga = (const float*)d_in[8];
  const float* be = (const float*)d_in[9];

  char* ws = (char*)d_ws;
  unsigned short* qb  = (unsigned short*)(ws + 0);          // 4 MB each
  unsigned short* kb  = (unsigned short*)(ws + 4194304);
  unsigned short* vb  = (unsigned short*)(ws + 8388608);
  unsigned short* wqT = (unsigned short*)(ws + 12582912);   // 512 KB each
  unsigned short* wkT = (unsigned short*)(ws + 13107200);
  unsigned short* wvT = (unsigned short*)(ws + 13631488);
  unsigned short* pwb = (unsigned short*)(ws + 14155776);
  unsigned short* qh  = (unsigned short*)(ws + 14680064);   // 4 MB each
  unsigned short* kh  = (unsigned short*)(ws + 18874368);
  unsigned short* vht = (unsigned short*)(ws + 23068672);
  unsigned short* ob  = (unsigned short*)(ws + 27262976);
  float*          y   = (float*)         (ws + 31457280);   // 8 MB

  float* out  = (float*)d_out;
  float* attn = out + (size_t)2 * S_LEN * D_MOD;            // after [B,S,D] output

  mha_pack_kernel<<<8192, 256, 0, stream>>>(q, k, v, wq, wk, wv, pw,
                                            qb, kb, vb, wqT, wkT, wvT, pwb);
  mha_qkv_proj_kernel<<<768, 256, 0, stream>>>(qb, kb, vb, wqT, wkT, wvT, qh, kh, vht);
  mha_attn_kernel<<<256, 256, 0, stream>>>(qh, kh, vht, attn, ob);
  mha_proj_kernel<<<256, 256, 0, stream>>>(ob, pwb, y);
  mha_ln_kernel<<<4096, 256, 0, stream>>>(y, q, pb, ga, be, out);
}
